// DRGNet_6287832121925
// MI455X (gfx1250) — compile-verified, matrix operand order checked by reading, not run on hardware
//
#include <hip/hip_runtime.h>
#include <hip/hip_bf16.h>

// ---------------- constants (fixed problem shape from the reference) ----------------
#define N_NODES   100000
#define N_EDGES   1600000
#define FEAT      64
#define NGRAPH    128
#define KTOP      30
#define TOTLAT    257           // 64*4 + 1
#define CH1       16
#define CH2       32
#define L1        30            // conv1 output length
#define LP        15            // after maxpool(2,2)
#define L2OUT     11            // after conv2 k=5 valid
#define DENSE     (L2OUT*CH2)   // 352
#define HIDDEN    128
#define NCLASS    5
#define MAXG      2048          // per-graph node cap for LDS sort (expected ~780)

typedef __attribute__((ext_vector_type(16))) __bf16 v16bf_t;
typedef __attribute__((ext_vector_type(8)))  float  v8f_t;

__device__ __forceinline__ float elu1(float v) {
    return v > 0.f ? v : (__expf(v) - 1.f);
}

// ---------------- edge message passing: agg[dst] += x[src] * ew ----------------
// 16 threads per edge, each moves one float4 chunk (fully coalesced 256B/edge),
// hardware fp32 atomics into L2-resident agg.
__global__ void __launch_bounds__(256)
drg_scatter(const float* __restrict__ x, const long long* __restrict__ src,
            const long long* __restrict__ dst, const float* __restrict__ ew,
            float* __restrict__ agg)
{
    long long idx = (long long)blockIdx.x * blockDim.x + threadIdx.x;
    long long e = idx >> 4;
    int c = (int)(idx & 15);
    if (e >= N_EDGES) return;
    long long s = src[e];
    long long d = dst[e];
    float w = ew[e];
    float4 v = ((const float4*)(x + s * FEAT))[c];
    float* out = agg + d * FEAT + c * 4;
    unsafeAtomicAdd(out + 0, v.x * w);
    unsafeAtomicAdd(out + 1, v.y * w);
    unsafeAtomicAdd(out + 2, v.z * w);
    unsafeAtomicAdd(out + 3, v.w * w);
}

// ---------------- WMMA fragment loader ----------------
// 16-bit A-matrix 16x32 layout (ISA 7.12.2): lane m = lane&15, half = lane>>4.
// VGPR0-3 hold K = half*8 + 0..7 packed; VGPR4-7 hold K = 16 + half*8 + 0..7.
// B (32x16, B[k][n] = W[n][k]) mirrors this per-lane K striping with n = lane&15,
// so a lane reads 2x8 contiguous fp32 from row n of W / row m of X and converts to bf16.
__device__ __forceinline__ v16bf_t drg_frag(const float* __restrict__ rowptr, int half)
{
    v16bf_t f;
    const int ka = half * 8;
    const int kb = 16 + half * 8;
#pragma unroll
    for (int i = 0; i < 8; ++i) f[i]     = (__bf16)rowptr[ka + i];
#pragma unroll
    for (int i = 0; i < 8; ++i) f[8 + i] = (__bf16)rowptr[kb + i];
    return f;
}

// ---------------- fused dual-GEMM GraphConv layer ----------------
// hout = ELU(agg @ Wrel^T + hin @ Wroot^T + b), all 64x64.
// Block = 4 waves; block covers rows [16*bid,16*bid+16), wave w covers cols [16w,16w+16).
// N_NODES % 16 == 0, so EXEC is all-ones at every WMMA (ISA requirement).
__global__ void __launch_bounds__(128)
drg_gconv_gemm(const float* __restrict__ agg, const float* __restrict__ hin,
               const float* __restrict__ w_rel, const float* __restrict__ w_root,
               const float* __restrict__ b_rel, float* __restrict__ hout)
{
    const int lane = threadIdx.x & 31;
    const int wave = threadIdx.x >> 5;
    const int half = lane >> 4;
    const int row0 = blockIdx.x * 16;
    const long long mrow = row0 + (lane & 15);
    const int ncol = (wave << 4) + (lane & 15);

    v8f_t acc = {};
#pragma unroll
    for (int k0 = 0; k0 < FEAT; k0 += 32) {
        v16bf_t a = drg_frag(agg + mrow * FEAT + k0, half);
        v16bf_t b = drg_frag(w_rel + (long long)ncol * FEAT + k0, half);
        acc = __builtin_amdgcn_wmma_f32_16x16x32_bf16(false, a, false, b,
                                                      (short)0, acc, false, false);
    }
#pragma unroll
    for (int k0 = 0; k0 < FEAT; k0 += 32) {
        v16bf_t a = drg_frag(hin + mrow * FEAT + k0, half);
        v16bf_t b = drg_frag(w_root + (long long)ncol * FEAT + k0, half);
        acc = __builtin_amdgcn_wmma_f32_16x16x32_bf16(false, a, false, b,
                                                      (short)0, acc, false, false);
    }
    // C/D layout: lane col = lane&15; VGPR r holds row r + 8*half.
    const float bias = b_rel[ncol];
#pragma unroll
    for (int i = 0; i < 8; ++i) {
        long long r = row0 + i + 8 * half;
        hout[r * FEAT + ncol] = elu1(acc[i] + bias);
    }
}

// ---------------- layer 5 (64 -> 1): score = ELU(agg.wrel5 + b + h4.wroot5) ----------------
__global__ void __launch_bounds__(256)
drg_layer5(const float* __restrict__ agg, const float* __restrict__ h4,
           const float* __restrict__ w_rel5, const float* __restrict__ w_root5,
           const float* __restrict__ b5, float* __restrict__ score)
{
    int i = blockIdx.x * blockDim.x + threadIdx.x;
    if (i >= N_NODES) return;
    float acc = b5[0];
    const float4* a4 = (const float4*)(agg + (long long)i * FEAT);
    const float4* h4p = (const float4*)(h4 + (long long)i * FEAT);
    const float4* wr = (const float4*)w_rel5;
    const float4* wo = (const float4*)w_root5;
#pragma unroll 4
    for (int c = 0; c < FEAT / 4; ++c) {
        float4 a = a4[c], h = h4p[c], r = wr[c], o = wo[c];
        acc += a.x * r.x + a.y * r.y + a.z * r.z + a.w * r.w;
        acc += h.x * o.x + h.y * o.y + h.z * o.z + h.w * o.w;
    }
    score[i] = elu1(acc);
}

// ---------------- per-graph node counts & starts ----------------
__global__ void __launch_bounds__(256)
drg_count(const long long* __restrict__ batch, int* __restrict__ counts)
{
    int i = blockIdx.x * blockDim.x + threadIdx.x;
    if (i < N_NODES) atomicAdd(&counts[(int)batch[i]], 1);
}

__global__ void drg_starts(const int* __restrict__ counts, int* __restrict__ starts)
{
    if (threadIdx.x == 0 && blockIdx.x == 0) {
        int s = 0;
        for (int g = 0; g < NGRAPH; ++g) { starts[g] = s; s += counts[g]; }
        starts[NGRAPH] = s;
    }
}

// ---------------- sort-pool: per graph, stable top-30 by score desc, gather 257 feats --------
// rank(i) = #{j: v[j] > v[i]} + #{j < i: v[j] == v[i]}  (== stable lexsort((-v, batch)))
__global__ void __launch_bounds__(256)
drg_sortpool(const float* __restrict__ score, const int* __restrict__ starts,
             const float* __restrict__ h1, const float* __restrict__ h2,
             const float* __restrict__ h3, const float* __restrict__ h4,
             float* __restrict__ pooled /* [NGRAPH][KTOP][TOTLAT], pre-zeroed */)
{
    __shared__ float sv[MAXG];
    __shared__ int sel[KTOP];
    const int g = blockIdx.x;
    const int beg = starts[g];
    int cnt = starts[g + 1] - beg;
    if (cnt > MAXG) cnt = MAXG;

    for (int i = threadIdx.x; i < cnt; i += blockDim.x) sv[i] = score[beg + i];
    __syncthreads();

    for (int i = threadIdx.x; i < cnt; i += blockDim.x) {
        const float vi = sv[i];
        int rank = 0;
        for (int j = 0; j < cnt; ++j) {
            float vj = sv[j];
            rank += (vj > vi) || (vj == vi && j < i);
        }
        if (rank < KTOP) sel[rank] = beg + i;
    }
    __syncthreads();

    const int nsel = cnt < KTOP ? cnt : KTOP;
    for (int t = threadIdx.x; t < nsel * TOTLAT; t += blockDim.x) {
        const int s = t / TOTLAT;
        const int d = t - s * TOTLAT;
        const long long node = sel[s];
        float v;
        if      (d < 64)  v = h1[node * FEAT + d];
        else if (d < 128) v = h2[node * FEAT + (d - 64)];
        else if (d < 192) v = h3[node * FEAT + (d - 128)];
        else if (d < 256) v = h4[node * FEAT + (d - 192)];
        else              v = score[node];
        pooled[((long long)g * KTOP + s) * TOTLAT + d] = v;
    }
}

// ---------------- conv1: per-slot linear 257 -> 16, ELU ----------------
__global__ void __launch_bounds__(256)
drg_conv1(const float* __restrict__ pooled, const float* __restrict__ w,
          const float* __restrict__ b, float* __restrict__ out /* [G][CH1][L1] */)
{
    int idx = blockIdx.x * blockDim.x + threadIdx.x;
    if (idx >= NGRAPH * CH1 * L1) return;
    const int s = idx % L1;
    const int o = (idx / L1) % CH1;
    const int g = idx / (L1 * CH1);
    const float* xp = pooled + ((long long)g * KTOP + s) * TOTLAT;
    const float* wr = w + (long long)o * TOTLAT;
    float acc = b[o];
    for (int d = 0; d < TOTLAT; ++d) acc += xp[d] * wr[d];
    out[((long long)g * CH1 + o) * L1 + s] = elu1(acc);
}

// ---------------- maxpool(2,2) fused into conv2 (16->32, k=5), ELU ----------------
__global__ void __launch_bounds__(256)
drg_conv2(const float* __restrict__ in /* [G][CH1][L1] */, const float* __restrict__ w,
          const float* __restrict__ b, float* __restrict__ out /* [G][CH2][L2OUT] */)
{
    int idx = blockIdx.x * blockDim.x + threadIdx.x;
    if (idx >= NGRAPH * CH2 * L2OUT) return;
    const int p = idx % L2OUT;
    const int c = (idx / L2OUT) % CH2;
    const int g = idx / (L2OUT * CH2);
    float acc = b[c];
#pragma unroll
    for (int i = 0; i < CH1; ++i) {
        const float* row = in + ((long long)g * CH1 + i) * L1;
        const float* wr = w + ((long long)c * CH1 + i) * 5;
#pragma unroll
        for (int kk = 0; kk < 5; ++kk) {
            const int t = p + kk;   // 0..14 pooled position
            acc += wr[kk] * fmaxf(row[2 * t], row[2 * t + 1]);
        }
    }
    out[((long long)g * CH2 + c) * L2OUT + p] = elu1(acc);
}

// ---------------- MLP ----------------
__global__ void __launch_bounds__(256)
drg_mlp1(const float* __restrict__ in /* [G][DENSE] */, const float* __restrict__ w,
         const float* __restrict__ b, float* __restrict__ out /* [G][HIDDEN] */)
{
    int idx = blockIdx.x * blockDim.x + threadIdx.x;
    if (idx >= NGRAPH * HIDDEN) return;
    const int j = idx & (HIDDEN - 1);
    const int g = idx >> 7;
    const float* xr = in + (long long)g * DENSE;
    const float* wr = w + (long long)j * DENSE;
    float acc = b[j];
    for (int d = 0; d < DENSE; ++d) acc += xr[d] * wr[d];
    out[idx] = elu1(acc);
}

__global__ void __launch_bounds__(256)
drg_mlp2(const float* __restrict__ in, const float* __restrict__ w,
         const float* __restrict__ b, float* __restrict__ out /* [G][NCLASS] */)
{
    int idx = blockIdx.x * blockDim.x + threadIdx.x;
    if (idx >= NGRAPH * NCLASS) return;
    const int c = idx % NCLASS;
    const int g = idx / NCLASS;
    float acc = b[c];
    const float* xr = in + (long long)g * HIDDEN;
    const float* wr = w + (long long)c * HIDDEN;
    for (int d = 0; d < HIDDEN; ++d) acc += xr[d] * wr[d];
    out[(long long)g * NCLASS + c] = acc;
}

// ---------------- launcher ----------------
extern "C" void kernel_launch(void* const* d_in, const int* in_sizes, int n_in,
                              void* d_out, int out_size, void* d_ws, size_t ws_size,
                              hipStream_t stream)
{
    // inputs (setup_inputs order; tuple members flattened in order)
    const float*     x      = (const float*)d_in[0];
    const long long* eidx   = (const long long*)d_in[1];   // int64 [2, E]
    const long long* batch  = (const long long*)d_in[2];   // int64 [N]
    const float*     ew     = (const float*)d_in[3];
    // d_in[4] = num_graphs (fixed 128)
    const float* w_rel[5]  = {(const float*)d_in[5],  (const float*)d_in[6],
                              (const float*)d_in[7],  (const float*)d_in[8],
                              (const float*)d_in[9]};
    const float* b_rel[5]  = {(const float*)d_in[10], (const float*)d_in[11],
                              (const float*)d_in[12], (const float*)d_in[13],
                              (const float*)d_in[14]};
    const float* w_root[5] = {(const float*)d_in[15], (const float*)d_in[16],
                              (const float*)d_in[17], (const float*)d_in[18],
                              (const float*)d_in[19]};
    const float* conv1_w = (const float*)d_in[20];
    const float* conv1_b = (const float*)d_in[21];
    const float* conv2_w = (const float*)d_in[22];
    const float* conv2_b = (const float*)d_in[23];
    const float* mlp_w1  = (const float*)d_in[24];
    const float* mlp_b1  = (const float*)d_in[25];
    const float* mlp_w2  = (const float*)d_in[26];
    const float* mlp_b2  = (const float*)d_in[27];

    const long long* src = eidx;
    const long long* dst = eidx + N_EDGES;

    // workspace carve-up (floats); ~134 MB total, fits in the 192 MB L2
    float* ws     = (float*)d_ws;
    float* agg    = ws;
    float* h1     = agg  + (size_t)N_NODES * FEAT;
    float* h2     = h1   + (size_t)N_NODES * FEAT;
    float* h3     = h2   + (size_t)N_NODES * FEAT;
    float* h4     = h3   + (size_t)N_NODES * FEAT;
    float* score  = h4   + (size_t)N_NODES * FEAT;
    float* pooled = score + N_NODES;
    float* c1out  = pooled + (size_t)NGRAPH * KTOP * TOTLAT;
    float* c2out  = c1out + (size_t)NGRAPH * CH1 * L1;
    float* m1out  = c2out + (size_t)NGRAPH * CH2 * L2OUT;
    int*   counts = (int*)(m1out + (size_t)NGRAPH * HIDDEN);
    int*   starts = counts + NGRAPH;

    const size_t aggBytes = (size_t)N_NODES * FEAT * sizeof(float);
    const int scatterBlocks = (N_EDGES * 16 + 255) / 256;   // 16 threads / edge
    const int gemmBlocks = N_NODES / 16;                    // 6250, exact
    const int nodeBlocks = (N_NODES + 255) / 256;

    const float* hin = x;
    float* houts[4] = {h1, h2, h3, h4};
    for (int l = 0; l < 4; ++l) {
        hipMemsetAsync(agg, 0, aggBytes, stream);
        drg_scatter<<<scatterBlocks, 256, 0, stream>>>(hin, src, dst, ew, agg);
        drg_gconv_gemm<<<gemmBlocks, 128, 0, stream>>>(agg, hin, w_rel[l], w_root[l],
                                                       b_rel[l], houts[l]);
        hin = houts[l];
    }

    // layer 5: 64 -> 1 score channel
    hipMemsetAsync(agg, 0, aggBytes, stream);
    drg_scatter<<<scatterBlocks, 256, 0, stream>>>(h4, src, dst, ew, agg);
    drg_layer5<<<nodeBlocks, 256, 0, stream>>>(agg, h4, w_rel[4], w_root[4],
                                               b_rel[4], score);

    // graph segmentation + sort-pool
    hipMemsetAsync(counts, 0, NGRAPH * sizeof(int), stream);
    drg_count<<<nodeBlocks, 256, 0, stream>>>(batch, counts);
    drg_starts<<<1, 32, 0, stream>>>(counts, starts);
    hipMemsetAsync(pooled, 0, (size_t)NGRAPH * KTOP * TOTLAT * sizeof(float), stream);
    drg_sortpool<<<NGRAPH, 256, 0, stream>>>(score, starts, h1, h2, h3, h4, pooled);

    // head
    drg_conv1<<<(NGRAPH * CH1 * L1 + 255) / 256, 256, 0, stream>>>(pooled, conv1_w,
                                                                   conv1_b, c1out);
    drg_conv2<<<(NGRAPH * CH2 * L2OUT + 255) / 256, 256, 0, stream>>>(c1out, conv2_w,
                                                                      conv2_b, c2out);
    drg_mlp1<<<(NGRAPH * HIDDEN + 255) / 256, 256, 0, stream>>>(c2out, mlp_w1,
                                                                mlp_b1, m1out);
    drg_mlp2<<<(NGRAPH * NCLASS + 255) / 256, 256, 0, stream>>>(m1out, mlp_w2,
                                                                mlp_b2, (float*)d_out);
}
